// MultiModalSelector_39178691674278
// MI455X (gfx1250) — matrix operand order repaired and checked by reading, not compiled, not run on hardware
//
#include <hip/hip_runtime.h>
#include <hip/hip_bf16.h>

typedef __attribute__((ext_vector_type(16))) __bf16 v16bf;
typedef __attribute__((ext_vector_type(8)))  __bf16 v8bf;
typedef __attribute__((ext_vector_type(8)))  float  v8f;
typedef __attribute__((ext_vector_type(4)))  float  v4f;

#define Bx 8
#define Nx 2048
#define Lx 64
#define Dx 128

// ================= WMMA micro-kernel helpers =================

template <bool NTLOAD>
__device__ __forceinline__ v16bf make_afrag_f32(const float* p, int ks, int half) {
    // A fragment, contiguous k order used here (B matches): lane<16 K[0..15]; lane>=16 K[16..31]
    const int k0 = ks * 32 + half * 16;
    v4f f0, f1, f2, f3;
    if constexpr (NTLOAD) {
        f0 = __builtin_nontemporal_load((const v4f*)(p + k0));
        f1 = __builtin_nontemporal_load((const v4f*)(p + k0 + 4));
        f2 = __builtin_nontemporal_load((const v4f*)(p + k0 + 8));
        f3 = __builtin_nontemporal_load((const v4f*)(p + k0 + 12));
    } else {
        f0 = *(const v4f*)(p + k0);
        f1 = *(const v4f*)(p + k0 + 4);
        f2 = *(const v4f*)(p + k0 + 8);
        f3 = *(const v4f*)(p + k0 + 12);
    }
    v16bf a;
#pragma unroll
    for (int i = 0; i < 4; ++i) {
        a[i]      = (__bf16)f0[i];
        a[4 + i]  = (__bf16)f1[i];
        a[8 + i]  = (__bf16)f2[i];
        a[12 + i] = (__bf16)f3[i];
    }
    return a;
}

__device__ __forceinline__ v16bf make_afrag_bf16(const __bf16* p, int ks, int half) {
    const int k0 = ks * 32 + half * 16;
    const v8bf c0 = *(const v8bf*)(p + k0);
    const v8bf c1 = *(const v8bf*)(p + k0 + 8);
    v16bf a;
#pragma unroll
    for (int i = 0; i < 8; ++i) { a[i] = c0[i]; a[8 + i] = c1[i]; }
    return a;
}

__device__ __forceinline__ void init_acc(v8f acc[8], const float* bias, int l16) {
#pragma unroll
    for (int nt = 0; nt < 8; ++nt) {
        const float bv = bias ? bias[nt * 16 + l16] : 0.f;
#pragma unroll
        for (int i = 0; i < 8; ++i) acc[nt][i] = bv;
    }
}

// acc[16x128] += A[16 x Kloop] @ B(columns from BT rows, stride BrowStride, offset Bkoff)
template <bool NTLOAD, bool ABF16>
__device__ __forceinline__ void wmma_accum(v8f acc[8], const void* aRowV, const __bf16* BT,
                                           int Kloop, int BrowStride, int Bkoff, int lane) {
    const int half = lane >> 4, l16 = lane & 15;
    for (int ks = 0; ks < (Kloop >> 5); ++ks) {
        v16bf a;
        if constexpr (ABF16) a = make_afrag_bf16((const __bf16*)aRowV, ks, half);
        else                 a = make_afrag_f32<NTLOAD>((const float*)aRowV, ks, half);
        const int kb0 = Bkoff + ks * 32 + half * 16;   // B in contiguous k order, same as the A fragment
        v16bf b[8];
#pragma unroll
        for (int nt = 0; nt < 8; ++nt)
            b[nt] = *(const v16bf*)(BT + (nt * 16 + l16) * BrowStride + kb0);
#pragma unroll
        for (int nt = 0; nt < 8; ++nt)
            acc[nt] = __builtin_amdgcn_wmma_f32_16x16x32_bf16(
                false, a, false, b[nt], (short)0, acc[nt], false, false);
    }
}

template <bool NT>
__device__ __forceinline__ void store_f32(const v8f acc[8], float* dst, int stride, int lane) {
    const int half = lane >> 4, l16 = lane & 15;
#pragma unroll
    for (int nt = 0; nt < 8; ++nt) {
        const int col = nt * 16 + l16;
#pragma unroll
        for (int v = 0; v < 8; ++v) {
            if constexpr (NT) __builtin_nontemporal_store(acc[nt][v], dst + (half * 8 + v) * stride + col);
            else              dst[(half * 8 + v) * stride + col] = acc[nt][v];
        }
    }
}

__device__ __forceinline__ void store_bf16(const v8f acc[8], __bf16* dst, int stride, int lane) {
    const int half = lane >> 4, l16 = lane & 15;
#pragma unroll
    for (int nt = 0; nt < 8; ++nt) {
        const int col = nt * 16 + l16;
#pragma unroll
        for (int v = 0; v < 8; ++v)
            dst[(half * 8 + v) * stride + col] = (__bf16)acc[nt][v];
    }
}

// ================= prep kernels =================

__global__ __launch_bounds__(256) void prep1_kernel(const float* __restrict__ vW,
                                                    const float* __restrict__ oW,
                                                    float* __restrict__ T1) {
    int idx = blockIdx.x * 256 + threadIdx.x;   // 16384
    int d = idx >> 7, j = idx & 127;
    float s = 0.f;
    for (int k = 0; k < 128; ++k) s += vW[d * 128 + k] * oW[k * 128 + j];
    T1[d * 128 + j] = s;
}

__global__ __launch_bounds__(256) void prep2_kernel(const float* __restrict__ xlW,
                                                    const float* __restrict__ T1,
                                                    const float* __restrict__ xlb,
                                                    const float* __restrict__ xpe,
                                                    const float* __restrict__ vb,
                                                    const float* __restrict__ oW,
                                                    __bf16* __restrict__ WcT,
                                                    float* __restrict__ bcomb) {
    int idx = blockIdx.x * 256 + threadIdx.x;   // 16384
    int d = idx >> 7, j = idx & 127;
    float s = 0.f;
    for (int k = 0; k < 128; ++k) s += xlW[d * 128 + k] * T1[k * 128 + j];
    WcT[j * 128 + d] = (__bf16)s;               // transposed [col][K]
    if (idx < 128) {
        float b = 0.f;
        for (int k = 0; k < 128; ++k)
            b += (xlb[k] + xpe[k]) * T1[k * 128 + idx] + vb[k] * oW[k * 128 + idx];
        bcomb[idx] = b;
    }
}

// src [K][128] f32 (row-major) -> dst [128][K] bf16
__global__ __launch_bounds__(256) void transpose_bf16_kernel(const float* __restrict__ src,
                                                             __bf16* __restrict__ dst, int K) {
    int idx = blockIdx.x * 256 + threadIdx.x;
    if (idx >= K * 128) return;
    int k = idx >> 7, c = idx & 127;
    dst[c * K + k] = (__bf16)src[idx];
}

__global__ __launch_bounds__(128) void biasq_kernel(const float* pb, const float* pp,
                                                    const float* sb, const float* sp,
                                                    const float* lb, const float* lp,
                                                    const float* xb, const float* xp,
                                                    float* bias_q) {
    int t = threadIdx.x;
    bias_q[t]       = pb[t] + pp[t];
    bias_q[128 + t] = sb[t] + sp[t];
    bias_q[256 + t] = lb[t] + lp[t];
    bias_q[384 + t] = xb[t] + xp[t];
}

__global__ __launch_bounds__(256) void xbar_kernel(const float* __restrict__ x,
                                                   float* __restrict__ xbar) {
    int idx = blockIdx.x * 256 + threadIdx.x;   // 262144
    int n = idx >> 7, d = idx & 127;
    float s = 0.f;
    for (int b = 0; b < Bx; ++b)
        s += x[(long long)b * (Nx * Lx * Dx) + (long long)n * (Lx * Dx) + d];
    xbar[idx] = s * 0.125f;
}

// ================= main fused GEMM: x_out = x @ Wcomb + bcomb =================
// 8 waves/block; each wave: 64 rows x 32 cols with the full B slice register-resident.
// Block covers 128 rows x 128 cols; grid 8192 covers all 1,048,576 rows exactly.
__global__ __launch_bounds__(256) void fused_xout_gemm(const float* __restrict__ x,
                                                       const __bf16* __restrict__ WcT,
                                                       const float* __restrict__ bcomb,
                                                       float* __restrict__ xout) {
    const int lane = threadIdx.x & 31;
    const int wave = threadIdx.x >> 5;
    const int half = lane >> 4, l16 = lane & 15;
    const int cp = wave & 3;                 // column pair: cols [cp*32, cp*32+32)
    const int rg = wave >> 2;                // row group of 64 rows
    const long long rowG = (long long)blockIdx.x * 128 + rg * 64;

    // B fragments (both column tiles, all 4 k-steps) stay in registers for the whole wave.
    v16bf bfrag[4][2];
#pragma unroll
    for (int ks = 0; ks < 4; ++ks)
#pragma unroll
        for (int j = 0; j < 2; ++j) {
            const int nt = cp * 2 + j;
            bfrag[ks][j] = *(const v16bf*)(WcT + (nt * 16 + l16) * 128 + ks * 32 + half * 16);
        }

    const int col0 = (cp * 2 + 0) * 16 + l16;
    const int col1 = (cp * 2 + 1) * 16 + l16;
    const float bias0 = bcomb[col0];
    const float bias1 = bcomb[col1];

#pragma unroll
    for (int rt = 0; rt < 4; ++rt) {
        const long long rowBase = rowG + rt * 16;
        const float* __restrict__ arow = x + (rowBase + l16) * Dx;
        v8f acc0, acc1;
#pragma unroll
        for (int i = 0; i < 8; ++i) { acc0[i] = bias0; acc1[i] = bias1; }
#pragma unroll
        for (int ks = 0; ks < 4; ++ks) {
            const v16bf a = make_afrag_f32<true>(arow, ks, half);
            acc0 = __builtin_amdgcn_wmma_f32_16x16x32_bf16(
                false, a, false, bfrag[ks][0], (short)0, acc0, false, false);
            acc1 = __builtin_amdgcn_wmma_f32_16x16x32_bf16(
                false, a, false, bfrag[ks][1], (short)0, acc1, false, false);
        }
        float* __restrict__ drow = xout + rowBase * Dx;
#pragma unroll
        for (int v = 0; v < 8; ++v) {
            __builtin_nontemporal_store(acc0[v], drow + (half * 8 + v) * Dx + col0);
            __builtin_nontemporal_store(acc1[v], drow + (half * 8 + v) * Dx + col1);
        }
    }
}

// ================= head: 16 n per block, wave m = modality m =================
__global__ __launch_bounds__(128) void head_kernel(
    const float* __restrict__ poi, const float* __restrict__ sat,
    const float* __restrict__ loc, const float* __restrict__ state,
    const float* __restrict__ q_b, const float* __restrict__ v_b,
    const float* __restrict__ imp_b, const float* __restrict__ unc_b1,
    const float* __restrict__ unc_W2, const float* __restrict__ unc_b2,
    const float* __restrict__ xbar, const float* __restrict__ bias_q,
    const __bf16* __restrict__ poi_WT, const __bf16* __restrict__ sat_WT,
    const __bf16* __restrict__ loc_WT, const __bf16* __restrict__ xl_WT,
    const __bf16* __restrict__ q_WT, const __bf16* __restrict__ v_WT,
    const __bf16* __restrict__ out_WT, const __bf16* __restrict__ imp_WT,
    const __bf16* __restrict__ unc_W1T,
    float* __restrict__ emb1, float* __restrict__ out3) {
    __shared__ __bf16 sQbf[4][16][128];    // modality embeddings q (bf16)
    __shared__ __bf16 sQybf[4][16][128];   // query
    __shared__ __bf16 sValbf[4][16][128];  // value
    __shared__ float  sImp[4][16], sUnc[4][16];

    const int lane = threadIdx.x & 31, wave = threadIdx.x >> 5;
    const int half = lane >> 4, l16 = lane & 15;
    const int m = wave, nbase = blockIdx.x * 16;
    v8f acc[8];

    // ---- phase A: q_m = modality_input @ W_m + (bias+pe) ----
    const float* aA; int KA; const __bf16* WA;
    switch (m) {
        case 0:  aA = poi  + (nbase + l16) * 64;  KA = 64;  WA = poi_WT; break;
        case 1:  aA = sat  + (nbase + l16) * 128; KA = 128; WA = sat_WT; break;
        case 2:  aA = loc  + (nbase + l16) * 32;  KA = 32;  WA = loc_WT; break;
        default: aA = xbar + (nbase + l16) * 128; KA = 128; WA = xl_WT;  break;
    }
    init_acc(acc, bias_q + m * 128, l16);
    wmma_accum<false, false>(acc, aA, WA, KA, KA, 0, lane);
    store_bf16(acc, &sQbf[m][0][0], 128, lane);

    // ---- query / value projections (H==1) ----
    init_acc(acc, q_b, l16);
    wmma_accum<false, true>(acc, &sQbf[m][l16][0], q_WT, 128, 128, 0, lane);
    store_bf16(acc, &sQybf[m][0][0], 128, lane);

    init_acc(acc, v_b, l16);
    wmma_accum<false, true>(acc, &sQbf[m][l16][0], v_WT, 128, 128, 0, lane);
    store_bf16(acc, &sValbf[m][0][0], 128, lane);
    __syncthreads();   // query_3 needed by all waves

    // ---- importance: sigmoid([query_m | query_3] @ imp_W[m] + b).mean(k) ----
    init_acc(acc, imp_b + m * 128, l16);
    wmma_accum<false, true>(acc, &sQybf[m][l16][0], imp_WT + m * 128 * 256, 128, 256, 0,   lane);
    wmma_accum<false, true>(acc, &sQybf[3][l16][0], imp_WT + m * 128 * 256, 128, 256, 128, lane);
#pragma unroll
    for (int v = 0; v < 8; ++v) {
        float p = 0.f;
#pragma unroll
        for (int nt = 0; nt < 8; ++nt) p += 1.f / (1.f + __expf(-acc[nt][v]));
#pragma unroll
        for (int off = 1; off < 16; off <<= 1) p += __shfl_xor(p, off);
        if (l16 == 0) sImp[m][half * 8 + v] = p * (1.f / 128.f);
    }

    // ---- uncertainty: sigmoid(relu(query_m @ unc_W1[m] + b1) . unc_W2[m] + b2) ----
    init_acc(acc, unc_b1 + m * 128, l16);
    wmma_accum<false, true>(acc, &sQybf[m][l16][0], unc_W1T + m * 128 * 128, 128, 128, 0, lane);
    {
        float w2c[8];
#pragma unroll
        for (int nt = 0; nt < 8; ++nt) w2c[nt] = unc_W2[m * 128 + nt * 16 + l16];
#pragma unroll
        for (int v = 0; v < 8; ++v) {
            float p = 0.f;
#pragma unroll
            for (int nt = 0; nt < 8; ++nt) {
                const float h = acc[nt][v] > 0.f ? acc[nt][v] : 0.f;
                p += h * w2c[nt];
            }
#pragma unroll
            for (int off = 1; off < 16; off <<= 1) p += __shfl_xor(p, off);
            if (l16 == 0) sUnc[m][half * 8 + v] = 1.f / (1.f + __expf(-(p + unc_b2[m])));
        }
    }

    // ---- emb1 = value @ out_W ----
    init_acc(acc, nullptr, l16);
    wmma_accum<false, true>(acc, &sValbf[m][l16][0], out_WT, 128, 128, 0, lane);
    store_f32<false>(acc, emb1 + (nbase * 4 + m) * 128, 512, lane);

    __syncthreads();
    // ---- gates + argmax one-hot + STE forward value ----
    if (threadIdx.x < 16) {
        const int n = nbase + threadIdx.x;
        float full[4];
#pragma unroll
        for (int mm = 0; mm < 4; ++mm) {
            const float logit = sImp[mm][threadIdx.x] / (1e-6f + sUnc[mm][threadIdx.x]);
            full[mm] = (1.f / (1.f + __expf(-logit * 2.0f))) * state[n * 4 + mm];  // /TAU, TAU=0.5
        }
        int am = 0;
#pragma unroll
        for (int mm = 1; mm < 4; ++mm) if (full[mm] > full[am]) am = mm;  // first max
#pragma unroll
        for (int mm = 0; mm < 4; ++mm)
            out3[n * 4 + mm] = (((mm == am) || (full[mm] > 0.8f)) ? 1.f : 0.f) * state[n * 4 + mm];
    }
}

// ================= launch =================
extern "C" void kernel_launch(void* const* d_in, const int* in_sizes, int n_in,
                              void* d_out, int out_size, void* d_ws, size_t ws_size,
                              hipStream_t stream) {
    const float* x      = (const float*)d_in[0];
    const float* poi    = (const float*)d_in[1];
    const float* sat    = (const float*)d_in[2];
    const float* loc    = (const float*)d_in[3];
    const float* state  = (const float*)d_in[4];
    const float* poi_W  = (const float*)d_in[5];
    const float* poi_b  = (const float*)d_in[6];
    const float* poi_pe = (const float*)d_in[7];
    const float* sat_W  = (const float*)d_in[8];
    const float* sat_b  = (const float*)d_in[9];
    const float* sat_pe = (const float*)d_in[10];
    const float* xl_W   = (const float*)d_in[11];
    const float* xl_b   = (const float*)d_in[12];
    const float* x_pe   = (const float*)d_in[13];
    const float* loc_W  = (const float*)d_in[14];
    const float* loc_b  = (const float*)d_in[15];
    const float* loc_pe = (const float*)d_in[16];
    const float* q_W    = (const float*)d_in[17];
    const float* q_b    = (const float*)d_in[18];
    const float* v_W    = (const float*)d_in[19];
    const float* v_b    = (const float*)d_in[20];
    const float* imp_W  = (const float*)d_in[21];
    const float* imp_b  = (const float*)d_in[22];
    const float* unc_W1 = (const float*)d_in[23];
    const float* unc_b1 = (const float*)d_in[24];
    const float* unc_W2 = (const float*)d_in[25];
    const float* unc_b2 = (const float*)d_in[26];
    const float* out_W  = (const float*)d_in[27];

    char* ws = (char*)d_ws;
    float*  T1      = (float*) (ws + 0);          // 64 KB
    __bf16* WcT     = (__bf16*)(ws + 65536);      // 32 KB
    float*  bcomb   = (float*) (ws + 98304);      // 512 B
    float*  xbar    = (float*) (ws + 98816);      // 1 MB
    __bf16* poi_WT  = (__bf16*)(ws + 1147392);    // 16 KB
    __bf16* sat_WT  = (__bf16*)(ws + 1163776);    // 32 KB
    __bf16* loc_WT  = (__bf16*)(ws + 1196544);    //  8 KB
    __bf16* xl_WT   = (__bf16*)(ws + 1204736);    // 32 KB
    __bf16* q_WT    = (__bf16*)(ws + 1237504);    // 32 KB
    __bf16* v_WT    = (__bf16*)(ws + 1270272);    // 32 KB
    __bf16* out_WT  = (__bf16*)(ws + 1303040);    // 32 KB
    __bf16* imp_WT  = (__bf16*)(ws + 1335808);    // 256 KB (4 x [128][256])
    __bf16* unc_W1T = (__bf16*)(ws + 1597952);    // 128 KB (4 x [128][128])
    float*  bias_q  = (float*) (ws + 1729024);    //  2 KB

    float* emb1 = (float*)d_out;                  // [2048,4,128]
    float* xout = (float*)d_out + 1048576;        // [8,2048,64,128]
    float* out3 = (float*)d_out + 135266304;      // [2048,4,1]

    prep1_kernel<<<64, 256, 0, stream>>>(v_W, out_W, T1);
    prep2_kernel<<<64, 256, 0, stream>>>(xl_W, T1, xl_b, x_pe, v_b, out_W, WcT, bcomb);
    xbar_kernel<<<1024, 256, 0, stream>>>(x, xbar);

    transpose_bf16_kernel<<<32, 256, 0, stream>>>(poi_W, poi_WT, 64);
    transpose_bf16_kernel<<<64, 256, 0, stream>>>(sat_W, sat_WT, 128);
    transpose_bf16_kernel<<<16, 256, 0, stream>>>(loc_W, loc_WT, 32);
    transpose_bf16_kernel<<<64, 256, 0, stream>>>(xl_W, xl_WT, 128);
    transpose_bf16_kernel<<<64, 256, 0, stream>>>(q_W, q_WT, 128);
    transpose_bf16_kernel<<<64, 256, 0, stream>>>(v_W, v_WT, 128);
    transpose_bf16_kernel<<<64, 256, 0, stream>>>(out_W, out_WT, 128);
    for (int m = 0; m < 4; ++m)
        transpose_bf16_kernel<<<128, 256, 0, stream>>>(imp_W + m * 32768, imp_WT + m * 32768, 256);
    for (int m = 0; m < 4; ++m)
        transpose_bf16_kernel<<<64, 256, 0, stream>>>(unc_W1 + m * 16384, unc_W1T + m * 16384, 128);
    biasq_kernel<<<1, 128, 0, stream>>>(poi_b, poi_pe, sat_b, sat_pe,
                                        loc_b, loc_pe, xl_b, x_pe, bias_q);

    fused_xout_gemm<<<8192, 256, 0, stream>>>(x, WcT, bcomb, xout);

    head_kernel<<<128, 128, 0, stream>>>(poi, sat, loc, state, q_b, v_b, imp_b, unc_b1,
                                         unc_W2, unc_b2, xbar, bias_q,
                                         poi_WT, sat_WT, loc_WT, xl_WT, q_WT, v_WT, out_WT,
                                         imp_WT, unc_W1T, emb1, out3);
}
